// E87SparseBlock_11416023073342
// MI455X (gfx1250) — matrix operand order repaired and checked by reading, not compiled, not run on hardware
//
#include <hip/hip_runtime.h>
#include <hip/hip_bf16.h>
#include <math.h>

// ---------------------------------------------------------------------------
// Model constants (from the reference)
// ---------------------------------------------------------------------------
#define DIM      1024
#define D_INNER  1024
#define N_STATE  32
#define N_BLOCKS 4
#define BATCH    16
#define T_LEN    2048
#define M_TOTAL  (BATCH * T_LEN)   // 32768 rows for all GEMMs
#define P_COLS   448               // 4 logits + 256 kv + 128 beta + 32 q + 28 pad

typedef __bf16 bf16_t;
typedef __attribute__((ext_vector_type(16))) __bf16 v16bf;
typedef __attribute__((ext_vector_type(8)))  __bf16 v8bf;
typedef __attribute__((ext_vector_type(8)))  float  v8f;
typedef __attribute__((ext_vector_type(4)))  int    i32x4;

// ---------------------------------------------------------------------------
// CDNA5 async global->LDS path (ASYNCcnt). Guarded: falls back to sync copies
// if the toolchain lacks the builtins. Builtin signature (from hipcc
// diagnostic): (int4 AS1*, int4 AS3*, imm offset, imm cpol), non-const.
// ---------------------------------------------------------------------------
#if __has_builtin(__builtin_amdgcn_global_load_async_to_lds_b128)
#define HAVE_ASYNC 1
#else
#define HAVE_ASYNC 0
#endif
#if __has_builtin(__builtin_amdgcn_s_wait_asynccnt)
#define HAVE_WAIT_ASYNC 1
#else
#define HAVE_WAIT_ASYNC 0
#endif

__device__ __forceinline__ void async_fill16(const bf16_t* g, bf16_t* l) {
#if HAVE_ASYNC
  __builtin_amdgcn_global_load_async_to_lds_b128(
      (__attribute__((address_space(1))) i32x4*)(uintptr_t)g,
      (__attribute__((address_space(3))) i32x4*)(unsigned)(uintptr_t)l,
      0, 0);
#else
  *(uint4*)l = *(const uint4*)g;            // sync fallback
#endif
}

template<int N>
__device__ __forceinline__ void wait_async() {
#if HAVE_ASYNC
#if HAVE_WAIT_ASYNC
  __builtin_amdgcn_s_wait_asynccnt(N);
#else
  asm volatile("s_wait_asynccnt %0" :: "i"(N));
#endif
#endif
}

// fast activations: v_rcp_f32 + v_exp_f32 instead of IEEE division chains
__device__ __forceinline__ float fast_sigmoid(float x) {
  return __builtin_amdgcn_rcpf(1.f + __expf(-x));
}
__device__ __forceinline__ float fast_tanh(float x) {
  return 1.f - 2.f * __builtin_amdgcn_rcpf(__expf(2.f * x) + 1.f);
}

// ---------------------------------------------------------------------------
// Kernel 1a: x (fp32) -> bf16, vectorized. ~200MB traffic ≈ 9us at 23.3 TB/s.
// ---------------------------------------------------------------------------
__global__ void convert_x_kernel(const float* __restrict__ xin,
                                 bf16_t* __restrict__ xb, int n4) {
  int i = blockIdx.x * blockDim.x + threadIdx.x;
  if (i >= n4) return;
  float4 v = ((const float4*)xin)[i];
  union { bf16_t h[4]; uint2 u; } o;
  o.h[0] = (bf16_t)v.x; o.h[1] = (bf16_t)v.y;
  o.h[2] = (bf16_t)v.z; o.h[3] = (bf16_t)v.w;
  *(uint2*)(xb + 4 * (size_t)i) = o.u;
}

// ---------------------------------------------------------------------------
// Kernel 1b: weights -> bf16; router/kv/beta/q fused into padded [448 x 1024].
// ---------------------------------------------------------------------------
__global__ void convert_weights_kernel(const float* __restrict__ W_in,
                                       const float* __restrict__ W_router,
                                       const float* __restrict__ W_kv,
                                       const float* __restrict__ W_beta,
                                       const float* __restrict__ W_q,
                                       const float* __restrict__ W_out,
                                       bf16_t* __restrict__ W_in_b,
                                       bf16_t* __restrict__ Wp_b,
                                       bf16_t* __restrict__ W_out_b) {
  int idx = blockIdx.x * blockDim.x + threadIdx.x;
  const int t1 = D_INNER * DIM;       // 1048576
  const int t2 = P_COLS * D_INNER;    // 458752
  const int t3 = DIM * N_STATE;       // 32768
  if (idx < t1) { W_in_b[idx] = (bf16_t)W_in[idx]; return; }
  idx -= t1;
  if (idx < t2) {
    int r = idx / D_INNER, k = idx % D_INNER;
    float v = 0.f;
    if      (r <   4) v = W_router[r * D_INNER + k];
    else if (r < 260) v = W_kv[(r - 4) * D_INNER + k];
    else if (r < 388) v = W_beta[(r - 260) * D_INNER + k];
    else if (r < 420) v = W_q[(r - 388) * D_INNER + k];
    Wp_b[idx] = (bf16_t)v;
    return;
  }
  idx -= t2;
  if (idx < t3) W_out_b[idx] = (bf16_t)W_out[idx];
}

// ---------------------------------------------------------------------------
// Kernel 2: bf16 WMMA GEMM  C[M,N] = A[M,K] * B[N,K]^T
//   - workgroup tile 128(M) x 64(N), 256 threads = 8 waves (4 along M, 2 N)
//   - per wave: 2x2 grid of v_wmma_f32_16x16x32_bf16 accumulators
//   - double-buffered LDS filled by async global->LDS DMA; stage s+2 is
//     issued after the consume barrier of stage s; s_wait_asynccnt 3 gates
//     reads (3 async fill instructions per thread per stage, in-order done)
// Requires M%128==0, N%64==0, K%32==0 (true for all call sites).
// ---------------------------------------------------------------------------
template<bool APPLY_SILU, bool OUT_BF16>
__global__ void __launch_bounds__(256)
gemm_bf16_wmma_kernel(const bf16_t* __restrict__ A, const bf16_t* __restrict__ B,
                      void* __restrict__ C_, int M, int N, int K) {
  __shared__ bf16_t ldsA[2][128 * 32];   // 2 x 8 KB
  __shared__ bf16_t ldsB[2][64 * 32];    // 2 x 4 KB
  const int tid  = threadIdx.x;
  const int lane = tid & 31;
  const int wave = tid >> 5;
  const int wm   = wave & 3;            // M sub-tile (32 rows)
  const int wn   = wave >> 2;           // N sub-tile (32 cols)
  const int mBase = blockIdx.x * 128;
  const int nBase = blockIdx.y * 64;

  v8f acc[2][2] = {};

  // 3 x 16B async fills per thread per stage (A: 2 chunks, B: 1 chunk)
  auto fill = [&](int buf, int k0) {
    #pragma unroll
    for (int c = tid; c < 512; c += 256) {     // A tile: 128 rows x 4 chunks
      int row = c >> 2, cc = c & 3;
      async_fill16(A + (size_t)(mBase + row) * K + k0 + cc * 8,
                   &ldsA[buf][row * 32 + cc * 8]);
    }
    {                                          // B tile: 64 rows x 4 chunks
      int row = tid >> 2, cc = tid & 3;
      async_fill16(B + (size_t)(nBase + row) * K + k0 + cc * 8,
                   &ldsB[buf][row * 32 + cc * 8]);
    }
  };

  const int S = K >> 5;                 // 32-wide K stages
  fill(0, 0);
  if (S > 1) fill(1, 32);

  const int arow_k = (lane < 16) ? 0 : 16;   // A frag: K 0-15 vs 16-31
  const int brow_k = (lane < 16) ? 0 : 16;   // B frag: K 0-15 vs 16-31

  for (int s = 0; s < S; ++s) {
    const int cur = s & 1;
    if (s + 1 < S) wait_async<3>();     // stage s done, stage s+1 in flight
    else           wait_async<0>();
    __syncthreads();

    v16bf afrag[2], bfrag[2];
    #pragma unroll
    for (int i = 0; i < 2; ++i) {
      int r = wm * 32 + i * 16 + (lane & 15);
      v8bf lo = *(const v8bf*)(&ldsA[cur][r * 32 + arow_k]);
      v8bf hi = *(const v8bf*)(&ldsA[cur][r * 32 + arow_k + 8]);
      afrag[i] = __builtin_shufflevector(lo, hi, 0,1,2,3,4,5,6,7,8,9,10,11,12,13,14,15);
    }
    #pragma unroll
    for (int j = 0; j < 2; ++j) {
      int r = wn * 32 + j * 16 + (lane & 15);
      bfrag[j] = *(const v16bf*)(&ldsB[cur][r * 32 + brow_k]);
    }
    #pragma unroll
    for (int i = 0; i < 2; ++i)
      #pragma unroll
      for (int j = 0; j < 2; ++j)
        acc[i][j] = __builtin_amdgcn_wmma_f32_16x16x32_bf16(
            false, afrag[i], false, bfrag[j], (short)0, acc[i][j], false, false);
    __syncthreads();                    // everyone done reading buffer `cur`

    if (s + 2 < S) fill(cur, (s + 2) * 32);   // refill behind the compute
  }

  // epilogue: C/D layout -> VGPR v holds rows v (lanes 0-15) / v+8 (16-31)
  const int rOff = (lane < 16) ? 0 : 8;
  const int cOff = lane & 15;
  #pragma unroll
  for (int i = 0; i < 2; ++i)
    #pragma unroll
    for (int j = 0; j < 2; ++j)
      #pragma unroll
      for (int v = 0; v < 8; ++v) {
        int gr = mBase + wm * 32 + i * 16 + v + rOff;
        int gc = nBase + wn * 32 + j * 16 + cOff;
        float val = acc[i][j][v];
        if constexpr (APPLY_SILU)
          val *= __builtin_amdgcn_rcpf(1.f + __expf(-val));   // silu via v_rcp
        if constexpr (OUT_BF16) ((bf16_t*)C_)[(size_t)gr * N + gc] = (bf16_t)val;
        else                    ((float*)C_)[(size_t)gr * N + gc]  = val;
      }
}

// ---------------------------------------------------------------------------
// Kernel 3: sequential scan. One block per batch (16 blocks x 512 threads).
// Thread = (nb = tid>>7, state row r = (tid>>2)&31, j-group jg = tid&3 of 8).
// Dot products close with 4-lane shuffle butterflies (wave32-native).
// ---------------------------------------------------------------------------
__global__ void __launch_bounds__(512)
sparse_block_scan_kernel(const float* __restrict__ P,
                         const float* __restrict__ b_beta,
                         bf16_t* __restrict__ outs_b,
                         float* __restrict__ S_final) {
  __shared__ float s_kn[N_BLOCKS][N_STATE];
  __shared__ float s_v [N_BLOCKS][N_STATE];
  __shared__ float s_q [N_STATE];
  __shared__ float s_sm[N_BLOCKS];
  __shared__ float s_w [N_BLOCKS];
  __shared__ float s_bo[N_BLOCKS][N_STATE];

  const int b   = blockIdx.x;
  const int tid = threadIdx.x;
  const int nb  = tid >> 7;
  const int r   = (tid >> 2) & 31;
  const int jg  = tid & 3;
  const int j0  = jg * 8;

  float S[8];
  #pragma unroll
  for (int u = 0; u < 8; ++u) S[u] = 0.f;
  const float bb = b_beta[nb * N_STATE + r];

  for (int t = 0; t < T_LEN; ++t) {
    const size_t m = (size_t)b * T_LEN + t;
    const float* Pm = P + m * P_COLS;

    if (tid < 128) {                       // waves 0..3: k-normalize + v
      const int knb = tid >> 5, kj = tid & 31;
      float kv = Pm[4 + knb * 64 + kj];
      float vv = Pm[4 + knb * 64 + 32 + kj];
      float ss = kv * kv;
      #pragma unroll
      for (int off = 16; off > 0; off >>= 1) ss += __shfl_xor(ss, off, 32);
      s_kn[knb][kj] = kv * __builtin_amdgcn_rcpf(sqrtf(ss) + 1e-6f);
      s_v[knb][kj]  = vv;
    } else if (tid < 160) {                // q
      s_q[tid - 128] = Pm[388 + (tid - 128)];
    } else if (tid == 160) {               // router softmax + top-2 weights
      float l[4] = {Pm[0], Pm[1], Pm[2], Pm[3]};
      float mx = fmaxf(fmaxf(l[0], l[1]), fmaxf(l[2], l[3]));
      float e[4], es = 0.f;
      for (int i = 0; i < 4; ++i) { e[i] = __expf(l[i] - mx); es += e[i]; }
      int i0 = 0;
      for (int i = 1; i < 4; ++i) if (l[i] > l[i0]) i0 = i;
      int i1 = (i0 == 0) ? 1 : 0;
      for (int i = 0; i < 4; ++i) if (i != i0 && l[i] > l[i1]) i1 = i;
      float inv_es = __builtin_amdgcn_rcpf(es);
      float uw[4], uws = 0.f;
      for (int i = 0; i < 4; ++i) {
        float sm = e[i] * inv_es;
        s_sm[i] = sm;
        uw[i] = (i == i0 || i == i1) ? sm : 0.f;
        uws += uw[i];
      }
      float inv_uws = __builtin_amdgcn_rcpf(uws + 1e-8f);
      for (int i = 0; i < 4; ++i) s_w[i] = uw[i] * inv_uws;
    }
    __syncthreads();

    const float beta = fast_sigmoid(Pm[260 + nb * N_STATE + r] + bb);
    const float w = s_w[nb];

    float p = 0.f;
    #pragma unroll
    for (int u = 0; u < 8; ++u) p += S[u] * s_kn[nb][j0 + u];
    p += __shfl_xor(p, 1, 4);
    p += __shfl_xor(p, 2, 4);
    const float delta = s_v[nb][r] - p;    // delta-rule error

    float sq = 0.f;
    #pragma unroll
    for (int u = 0; u < 8; ++u) {
      float kn = s_kn[nb][j0 + u];
      float su = fast_tanh(beta * S[u] + delta * kn);
      S[u] = (1.f - w) * S[u] + w * su;
      sq += S[u] * s_q[j0 + u];
    }
    sq += __shfl_xor(sq, 1, 4);
    sq += __shfl_xor(sq, 2, 4);
    const float bo = sq * sq * fast_sigmoid(sq);   // Sq * silu(Sq)
    if (jg == 0) s_bo[nb][r] = bo;
    __syncthreads();

    if (tid < 32) {
      float o = 0.f;
      #pragma unroll
      for (int i = 0; i < 4; ++i) o += s_sm[i] * s_bo[i][tid];
      outs_b[m * N_STATE + tid] = (bf16_t)o;
    }
    __syncthreads();                       // protect shared state for next t
  }

  // S_final: [B, N_BLOCKS, 32, 32] fp32
  size_t base = (((size_t)b * N_BLOCKS + nb) * N_STATE + r) * N_STATE + j0;
  #pragma unroll
  for (int u = 0; u < 8; ++u) S_final[base + u] = S[u];
}

// ---------------------------------------------------------------------------
// Launch pipeline
// ---------------------------------------------------------------------------
extern "C" void kernel_launch(void* const* d_in, const int* in_sizes, int n_in,
                              void* d_out, int out_size, void* d_ws, size_t ws_size,
                              hipStream_t stream) {
  (void)in_sizes; (void)n_in; (void)out_size; (void)ws_size;
  const float* x        = (const float*)d_in[0];
  const float* W_in     = (const float*)d_in[1];
  const float* W_router = (const float*)d_in[2];
  const float* W_kv     = (const float*)d_in[3];
  const float* W_beta   = (const float*)d_in[4];
  const float* b_beta   = (const float*)d_in[5];
  const float* W_q      = (const float*)d_in[6];
  const float* W_out    = (const float*)d_in[7];

  // workspace layout (bytes), ~190 MB total
  char* ws = (char*)d_ws;
  bf16_t* W_in_b  = (bf16_t*)(ws + 0);           //  2,097,152
  bf16_t* Wp_b    = (bf16_t*)(ws + 2097152);     //    917,504
  bf16_t* W_out_b = (bf16_t*)(ws + 3014656);     //     65,536
  bf16_t* x_b     = (bf16_t*)(ws + 3080192);     // 67,108,864
  bf16_t* xproj_b = (bf16_t*)(ws + 70189056);    // 67,108,864
  float*  P       = (float*) (ws + 137297920);   // 58,720,256
  bf16_t* outs_b  = (bf16_t*)(ws + 196018176);   //  2,097,152

  float* y       = (float*)d_out;
  float* S_final = y + (size_t)BATCH * T_LEN * DIM;

  // 1a) x -> bf16
  {
    int n4 = (M_TOTAL * DIM) / 4;
    convert_x_kernel<<<(n4 + 255) / 256, 256, 0, stream>>>(x, x_b, n4);
  }
  // 1b) weights -> bf16 (router/kv/beta/q fused into Wp)
  {
    int total = D_INNER * DIM + P_COLS * D_INNER + DIM * N_STATE;
    convert_weights_kernel<<<(total + 255) / 256, 256, 0, stream>>>(
        W_in, W_router, W_kv, W_beta, W_q, W_out, W_in_b, Wp_b, W_out_b);
  }
  // 2) x_proj = silu(x @ W_in^T)   [bf16 -> bf16]   68.7 GFLOP
  gemm_bf16_wmma_kernel<true, true>
      <<<dim3(M_TOTAL / 128, D_INNER / 64), 256, 0, stream>>>(
          x_b, W_in_b, xproj_b, M_TOTAL, D_INNER, DIM);
  // 3) P = x_proj @ Wp^T           [bf16 -> fp32]   ~29 GFLOP
  gemm_bf16_wmma_kernel<false, false>
      <<<dim3(M_TOTAL / 128, P_COLS / 64), 256, 0, stream>>>(
          xproj_b, Wp_b, P, M_TOTAL, P_COLS, DIM);
  // 4) sequential recurrence (latency-bound)
  sparse_block_scan_kernel<<<BATCH, 512, 0, stream>>>(P, b_beta, outs_b, S_final);
  // 5) y = outs @ W_out^T          [K=32, single WMMA slice]
  gemm_bf16_wmma_kernel<false, false>
      <<<dim3(M_TOTAL / 128, DIM / 64), 256, 0, stream>>>(
          outs_b, W_out_b, y, M_TOTAL, DIM, N_STATE);
}
